// StochasticDurationPredictor_9354438771258
// MI455X (gfx1250) — matrix operand order repaired and checked by reading, not compiled, not run on hardware
//
#include <hip/hip_runtime.h>
#include <math.h>

#define DEVFN __device__ __forceinline__

namespace {

constexpr int   B_   = 32;
constexpr int   C_   = 192;
constexpr int   T_   = 2048;
constexpr float TAILF = 5.0f;
constexpr float LOG2PI_F = 1.8378770664093453f;
constexpr float MIN_BW = 1e-3f;
constexpr float MIN_D  = 1e-3f;
constexpr float INV_SQRT_C = 0.07216878364870322f; // 1/sqrt(192)

typedef __attribute__((ext_vector_type(16))) __bf16 v16bf;
typedef __attribute__((ext_vector_type(8)))  float  v8f;

union UV16 { unsigned short u[16]; v16bf v; };

DEVFN unsigned short f2bfu(float f) {            // f32 -> bf16 bits, round-nearest-even
  unsigned int x = __builtin_bit_cast(unsigned int, f);
  x += 0x7fffu + ((x >> 16) & 1u);
  return (unsigned short)(x >> 16);
}
DEVFN float geluf(float x) { return 0.5f * x * (1.0f + erff(x * 0.7071067811865476f)); }
DEVFN float logsigf(float x) { return x < 0.0f ? (x - log1pf(expf(x))) : (-log1pf(expf(-x))); }
DEVFN float softplusf(float x) { return x > 20.0f ? x : log1pf(expf(x)); }

// ---- WMMA operand loaders (contiguous k order, same for A and B; wave32) -----------
// A (16x32 bf16): lanes 0-15 row M=lane, K in {0..15}; lanes 16-31 same
// rows, K in {16..31}.
DEVFN v16bf load_a_tile(const unsigned short* Wm, int ld, int mi, int kc, int lane) {
  const unsigned short* p = Wm + (size_t)(mi * 16 + (lane & 15)) * ld + kc * 32 + ((lane & 16) ? 16 : 0);
  UV16 t;
#pragma unroll
  for (int e = 0; e < 8; ++e) t.u[e] = p[e];
#pragma unroll
  for (int e = 0; e < 8; ++e) t.u[8 + e] = p[8 + e];
  return t.v;
}
// B (32x16 bf16): lane n<16 holds column N=n, K=kc*32+0..15 (contiguous);
// lane n+16 holds column N=n, K=kc*32+16..31.
DEVFN v16bf load_b_tile(const unsigned short (*ys)[200], int ni, int kc, int lane) {
  const unsigned short* p = &ys[ni * 16 + (lane & 15)][kc * 32 + ((lane & 16) ? 16 : 0)];
  UV16 t;
#pragma unroll
  for (int e = 0; e < 16; ++e) t.u[e] = p[e];
  return t.v;
}

// ---- per-column (over 192 channels) mean / rstd; 384-thread blocks --------
DEVFN void colstats384(const float (*zs)[193], float* red, float* mu, float* rs, int tid) {
  if (tid < 256) {
    int p = tid >> 5, n = tid & 31;
    float s = 0.f, q = 0.f;
    int c0 = p * 24;
#pragma unroll 4
    for (int c = c0; c < c0 + 24; ++c) { float v = zs[n][c]; s += v; q += v * v; }
    red[p * 32 + n] = s;
    red[256 + p * 32 + n] = q;
  }
  __syncthreads();
  if (tid < 32) {
    float S = 0.f, Q = 0.f;
#pragma unroll
    for (int pp = 0; pp < 8; ++pp) { S += red[pp * 32 + tid]; Q += red[256 + pp * 32 + tid]; }
    float m = S * (1.0f / 192.0f);
    float v = Q * (1.0f / 192.0f) - m * m;
    mu[tid] = m;
    rs[tid] = rsqrtf(v + 1e-5f);
  }
  __syncthreads();
}

// ============================== kernels ====================================

// z[b][ch][t] = m[ch] + exp(logs[ch]) * src[b][ch][t]   (src may alias z)
__global__ __launch_bounds__(256) void k_flow_init(const float* __restrict__ src, float* __restrict__ z,
                                                   const float* __restrict__ m, const float* __restrict__ logs) {
  size_t i = (size_t)blockIdx.x * 256 + threadIdx.x;
  if (i >= (size_t)B_ * 2 * T_) return;
  int ch = (int)((i / T_) & 1);
  z[i] = m[ch] + expf(logs[ch]) * src[i];
}

__global__ void k_acc_logs(float* acc, const float* __restrict__ logs, int add) {
  int b = threadIdx.x;
  if (b >= B_) return;
  float s = (logs[0] + logs[1]) * (float)T_;
  acc[b] = add ? acc[b] + s : s;
}

// Convert conv-flow weights to bf16 (pw_w x3, proj_w padded 29->32 rows)
__global__ __launch_bounds__(256) void k_prep(const float* __restrict__ pw0, const float* __restrict__ pw1,
                                              const float* __restrict__ pw2, const float* __restrict__ prw,
                                              const float* __restrict__ prb,
                                              unsigned short* __restrict__ wpw, unsigned short* __restrict__ wpr,
                                              float* __restrict__ wprb) {
  int i = blockIdx.x * 256 + threadIdx.x;
  const int n1 = C_ * C_;
  if (i < 3 * n1) {
    const float* s = (i < n1) ? pw0 : ((i < 2 * n1) ? pw1 : pw2);
    wpw[i] = f2bfu(s[i % n1]);
  } else if (i < 3 * n1 + 32 * C_) {
    int j = i - 3 * n1;
    int r = j / C_;
    wpr[j] = f2bfu(r < 29 ? prw[j] : 0.0f);
  } else if (i < 3 * n1 + 32 * C_ + 32) {
    int r = i - 3 * n1 - 32 * C_;
    wprb[r] = (r < 29) ? prb[r] : 0.0f;
  }
}

template <int D, bool PRE>
struct SmemDds {
  static constexpr int WDv = 32 + 2 * D;
  union {
    float          xs[PRE ? 1 : C_][PRE ? 1 : WDv];  // haloed tile (non-PRE, phase 1)
    unsigned short ysb[32][200];                     // bf16 B operand [col][chan]
  } u;
  float zs[32][193];                                 // f32 scratch [col][chan]
  float red[512];
  float mu[32];
  float rs[32];
  float zrow[PRE ? WDv : 1];                         // PRE: haloed z0 row
  float mrow[PRE ? WDv : 1];                         // PRE: validity mask
};
static_assert(sizeof(SmemDds<9, false>) <= 65536, "LDS budget");
static_assert(sizeof(SmemDds<1, true>)  <= 65536, "LDS budget");

// One fused dds layer: [optional pre 1x1 expansion +] depthwise dilated conv +
// LN + GELU + 192x192 WMMA GEMM + LN + GELU + residual.
// Tile: 32 time steps, one batch.  384 threads = 12 wave32, one M-tile/wave.
template <int D, bool PRE>
__global__ __launch_bounds__(384) void k_dds(const float* __restrict__ hin, float* __restrict__ hout,
                                             const float* __restrict__ sep_w, const float* __restrict__ sep_b,
                                             const float* __restrict__ g1, const float* __restrict__ b1,
                                             const unsigned short* __restrict__ pwW, const float* __restrict__ pw_b,
                                             const float* __restrict__ g2, const float* __restrict__ b2,
                                             const float* __restrict__ pre_w, const float* __restrict__ pre_b) {
  __shared__ SmemDds<D, PRE> sm;
  const int tid = threadIdx.x;
  const int t0  = blockIdx.x * 32;
  const int b   = blockIdx.y;
  constexpr int WD = 32 + 2 * D;
  const float* hb = hin + (size_t)b * C_ * T_;       // meaningful when !PRE
  const float* zb = hin + ((size_t)b * 2) * T_;      // meaningful when PRE (ch 0)

  if constexpr (PRE) {
    // phase 1: only the haloed z0 row is needed; h is synthesized on the fly.
    if (tid < WD) {
      int t = t0 - D + tid;
      bool ok = (t >= 0) && (t < T_);
      sm.zrow[tid] = ok ? zb[t] : 0.0f;
      sm.mrow[tid] = ok ? 1.0f : 0.0f;
    }
    __syncthreads();
    for (int e = tid; e < C_ * 32; e += 384) {
      int c = e >> 5, j = e & 31;
      float pw = pre_w[c], pb = pre_b[c];
      float h0 = sm.mrow[j]         * (pw * sm.zrow[j]         + pb);
      float h1 = sm.mrow[j + D]     * (pw * sm.zrow[j + D]     + pb);
      float h2 = sm.mrow[j + 2 * D] * (pw * sm.zrow[j + 2 * D] + pb);
      sm.zs[j][c] = sep_w[c * 3 + 0] * h0 + sep_w[c * 3 + 1] * h1 +
                    sep_w[c * 3 + 2] * h2 + sep_b[c];
    }
  } else {
    // phase 1: async copy of haloed tile global -> LDS (no VGPR staging)
    for (int e = tid; e < C_ * WD; e += 384) {
      int c = e / WD, j = e % WD;
      int t = t0 - D + j;
      if (t >= 0 && t < T_) {
        unsigned lds = (unsigned)(unsigned long long)&sm.u.xs[c][j];
        const float* g = hb + (size_t)c * T_ + t;
        asm volatile("global_load_async_to_lds_b32 %0, %1, off" :: "v"(lds), "v"(g) : "memory");
      } else {
        sm.u.xs[c][j] = 0.0f;
      }
    }
    asm volatile("s_wait_asynccnt 0x0" ::: "memory");
    __syncthreads();
    // depthwise dilated conv -> zs[col][chan]
    for (int e = tid; e < C_ * 32; e += 384) {
      int c = e >> 5, j = e & 31;
      sm.zs[j][c] = sep_w[c * 3 + 0] * sm.u.xs[c][j] +
                    sep_w[c * 3 + 1] * sm.u.xs[c][j + D] +
                    sep_w[c * 3 + 2] * sm.u.xs[c][j + 2 * D] + sep_b[c];
    }
  }
  __syncthreads();

  colstats384(sm.zs, sm.red, sm.mu, sm.rs, tid);

  // LN1 + GELU -> bf16 B operand (overwrites xs; residual re-derived later)
  for (int e = tid; e < C_ * 32; e += 384) {
    int c = e >> 5, j = e & 31;
    float v = (sm.zs[j][c] - sm.mu[j]) * sm.rs[j] * g1[c] + b1[c];
    sm.u.ysb[j][c] = f2bfu(geluf(v));
  }
  if constexpr (!PRE) {
    if (tid < C_) __builtin_prefetch(hb + (size_t)tid * T_ + t0, 0, 1);  // residual tile
  }
  __syncthreads();

  // 192x192 x (192x32) GEMM: 12 waves, wave = one M-tile; A fragment loaded
  // once per K-step and reused for both N-tiles (two back-to-back WMMAs).
  const int wv = tid >> 5, lane = tid & 31;
  {
    const int mi = wv;  // 0..11
    v8f acc0, acc1;
#pragma unroll
    for (int r = 0; r < 8; ++r) {
      float bias = pw_b[mi * 16 + r + ((lane & 16) ? 8 : 0)];
      acc0[r] = bias;
      acc1[r] = bias;
    }
#pragma unroll
    for (int kc = 0; kc < 6; ++kc) {
      v16bf a  = load_a_tile(pwW, C_, mi, kc, lane);
      v16bf b0 = load_b_tile(sm.u.ysb, 0, kc, lane);
      v16bf b1v = load_b_tile(sm.u.ysb, 1, kc, lane);
      acc0 = __builtin_amdgcn_wmma_f32_16x16x32_bf16(false, a, false, b0,  (short)0, acc0, false, false);
      acc1 = __builtin_amdgcn_wmma_f32_16x16x32_bf16(false, a, false, b1v, (short)0, acc1, false, false);
    }
    int n = lane & 15;
    int mo = mi * 16 + ((lane & 16) ? 8 : 0);
#pragma unroll
    for (int r = 0; r < 8; ++r) sm.zs[n][mo + r]      = acc0[r];
#pragma unroll
    for (int r = 0; r < 8; ++r) sm.zs[16 + n][mo + r] = acc1[r];
  }
  __syncthreads();

  colstats384(sm.zs, sm.red, sm.mu, sm.rs, tid);

  // LN2 + GELU + residual -> global
  for (int e = tid; e < C_ * 32; e += 384) {
    int c = e >> 5, j = e & 31;
    float v = (sm.zs[j][c] - sm.mu[j]) * sm.rs[j] * g2[c] + b2[c];
    float res;
    if constexpr (PRE) res = pre_w[c] * sm.zrow[D + j] + pre_b[c];
    else               res = hb[(size_t)c * T_ + t0 + j];
    hout[(size_t)b * C_ * T_ + (size_t)c * T_ + t0 + j] = res + geluf(v);
  }
}

// Fused: 32x192 WMMA projection + rational-quadratic spline on ch1 + channel
// flip (in place on z) + per-block log-det partial sum.
__global__ __launch_bounds__(256) void k_proj_rqs(const float* __restrict__ hin, float* __restrict__ z,
                                                  const unsigned short* __restrict__ prW,
                                                  const float* __restrict__ prB,
                                                  float* __restrict__ lad_part) {
  __shared__ unsigned short bsb[32][200];
  __shared__ float hc[32][33];
  __shared__ float sred[32];
  const int tid = threadIdx.x;
  const int t0  = blockIdx.x * 32;
  const int b   = blockIdx.y;
  const float* hb = hin + (size_t)b * C_ * T_;

  for (int e = tid; e < C_ * 32; e += 256) {
    int c = e >> 5, j = e & 31;
    bsb[j][c] = f2bfu(hb[(size_t)c * T_ + t0 + j]);
  }
  __syncthreads();

  const int wv = tid >> 5, lane = tid & 31;
  if (wv < 2) {
    const int mi = wv;
    v8f acc0, acc1;
#pragma unroll
    for (int r = 0; r < 8; ++r) {
      float bias = prB[mi * 16 + r + ((lane & 16) ? 8 : 0)];
      acc0[r] = bias;
      acc1[r] = bias;
    }
#pragma unroll
    for (int kc = 0; kc < 6; ++kc) {
      v16bf a  = load_a_tile(prW, C_, mi, kc, lane);
      v16bf b0 = load_b_tile(bsb, 0, kc, lane);
      v16bf b1v = load_b_tile(bsb, 1, kc, lane);
      acc0 = __builtin_amdgcn_wmma_f32_16x16x32_bf16(false, a, false, b0,  (short)0, acc0, false, false);
      acc1 = __builtin_amdgcn_wmma_f32_16x16x32_bf16(false, a, false, b1v, (short)0, acc1, false, false);
    }
    int n = lane & 15;
    int mo = mi * 16 + ((lane & 16) ? 8 : 0);
#pragma unroll
    for (int r = 0; r < 8; ++r) hc[n][mo + r]      = acc0[r];
#pragma unroll
    for (int r = 0; r < 8; ++r) hc[16 + n][mo + r] = acc1[r];
  }
  __syncthreads();

  if (tid < 32) {
    const int n = tid, t = t0 + n;
    float x0 = z[((size_t)b * 2) * T_ + t];
    float x1 = z[((size_t)b * 2 + 1) * T_ + t];

    float cwv[11], chv[11], dvv[11], e10[10];
    // widths
    float mx = -1e30f;
#pragma unroll
    for (int k = 0; k < 10; ++k) { float v = hc[n][k] * INV_SQRT_C; e10[k] = v; mx = fmaxf(mx, v); }
    float se = 0.f;
#pragma unroll
    for (int k = 0; k < 10; ++k) { e10[k] = expf(e10[k] - mx); se += e10[k]; }
    float inv = 1.f / se, cum = 0.f;
    cwv[0] = -TAILF;
#pragma unroll
    for (int k = 0; k < 10; ++k) { cum += MIN_BW + 0.99f * e10[k] * inv; cwv[k + 1] = 2.f * TAILF * cum - TAILF; }
    cwv[10] = TAILF;
    // heights
    mx = -1e30f;
#pragma unroll
    for (int k = 0; k < 10; ++k) { float v = hc[n][10 + k] * INV_SQRT_C; e10[k] = v; mx = fmaxf(mx, v); }
    se = 0.f;
#pragma unroll
    for (int k = 0; k < 10; ++k) { e10[k] = expf(e10[k] - mx); se += e10[k]; }
    inv = 1.f / se; cum = 0.f;
    chv[0] = -TAILF;
#pragma unroll
    for (int k = 0; k < 10; ++k) { cum += MIN_BW + 0.99f * e10[k] * inv; chv[k + 1] = 2.f * TAILF * cum - TAILF; }
    chv[10] = TAILF;
    // knot derivatives (padded ends -> exactly 1.0)
    dvv[0] = 1.0f; dvv[10] = 1.0f;
#pragma unroll
    for (int k = 1; k < 10; ++k) dvv[k] = MIN_D + softplusf(hc[n][20 + k - 1]);

    bool inside = (x1 >= -TAILF) && (x1 <= TAILF);
    float xin = fminf(fmaxf(x1, -TAILF), TAILF);
    int id = 0;
#pragma unroll
    for (int k = 1; k < 11; ++k) id += (xin >= cwv[k]) ? 1 : 0;
    if (id > 9) id = 9;

    float ibw = cwv[id + 1] - cwv[id];
    float ih  = chv[id + 1] - chv[id];
    float idl = ih / ibw;
    float d0 = dvv[id], d1 = dvv[id + 1];
    float th = (xin - cwv[id]) / ibw;
    float tom = th * (1.f - th);
    float num = ih * (idl * th * th + d0 * tom);
    float den = idl + (d0 + d1 - 2.f * idl) * tom;
    float out = chv[id] + num / den;
    float dnum = idl * idl * (d1 * th * th + 2.f * idl * tom + d0 * (1.f - th) * (1.f - th));
    float lad = logf(dnum) - 2.f * logf(den);

    float y1 = inside ? out : x1;
    // fused channel flip
    z[((size_t)b * 2) * T_ + t]     = y1;
    z[((size_t)b * 2 + 1) * T_ + t] = x0;
    sred[n] = inside ? lad : 0.0f;
  }
  __syncthreads();
  if (tid == 0) {
    float s = 0.f;
#pragma unroll
    for (int k = 0; k < 32; ++k) s += sred[k];
    lad_part[b * 64 + blockIdx.x] = s;
  }
}

__global__ void k_reduce_lad(const float* __restrict__ part, float* acc, int n) {
  __shared__ float s[64];
  int b = blockIdx.x, tid = threadIdx.x;
  float v = 0.f;
  for (int i = tid; i < n; i += 64) v += part[b * n + i];
  s[tid] = v;
  __syncthreads();
  for (int st = 32; st; st >>= 1) { if (tid < st) s[tid] += s[tid + st]; __syncthreads(); }
  if (tid == 0) acc[b] += s[0];
}

__global__ __launch_bounds__(256) void k_mid_elem(float* __restrict__ z, const float* __restrict__ w,
                                                  const float* __restrict__ e_q,
                                                  float* __restrict__ p_ls, float* __restrict__ p_eq,
                                                  float* __restrict__ p_y0) {
  __shared__ float s[256];
  int b = blockIdx.y, tid = threadIdx.x;
  int t = blockIdx.x * 256 + tid;
  float zu = z[((size_t)b * 2) * T_ + t];
  float wv = w[(size_t)b * T_ + t];
  float e0 = e_q[((size_t)b * 2) * T_ + t];
  float e1 = e_q[((size_t)b * 2 + 1) * T_ + t];
  float u  = 1.f / (1.f + expf(-zu));
  float z0 = wv - u;
  float y0 = logf(fmaxf(z0, 1e-5f));
  z[((size_t)b * 2) * T_ + t] = y0;

  float v0 = logsigf(zu) + logsigf(-zu);
  float v1 = -LOG2PI_F - 0.5f * (e0 * e0 + e1 * e1);
  float v2 = -y0;
  float* outs[3] = {p_ls, p_eq, p_y0};
  float vals[3] = {v0, v1, v2};
#pragma unroll
  for (int q = 0; q < 3; ++q) {
    s[tid] = vals[q];
    __syncthreads();
    for (int st = 128; st; st >>= 1) { if (tid < st) s[tid] += s[tid + st]; __syncthreads(); }
    if (tid == 0) outs[q][b * gridDim.x + blockIdx.x] = s[0];
    __syncthreads();
  }
}

__global__ void k_mid_fin(const float* __restrict__ p_ls, const float* __restrict__ p_eq,
                          const float* __restrict__ p_y0, float* acc_ld, float* logq, int n) {
  int b = threadIdx.x;
  if (b >= B_) return;
  float sls = 0.f, seq = 0.f, sy = 0.f;
  for (int i = 0; i < n; ++i) { sls += p_ls[b * n + i]; seq += p_eq[b * n + i]; sy += p_y0[b * n + i]; }
  float logdet_q = acc_ld[b] + sls;
  logq[b] = seq - logdet_q;
  acc_ld[b] = sy;   // becomes logdet_tot accumulator for chain 2
}

__global__ __launch_bounds__(256) void k_nll_elem(const float* __restrict__ z, float* __restrict__ part) {
  __shared__ float s[256];
  int b = blockIdx.y, tid = threadIdx.x;
  int t = blockIdx.x * 256 + tid;
  float z0 = z[((size_t)b * 2) * T_ + t];
  float z1 = z[((size_t)b * 2 + 1) * T_ + t];
  s[tid] = LOG2PI_F + 0.5f * (z0 * z0 + z1 * z1);
  __syncthreads();
  for (int st = 128; st; st >>= 1) { if (tid < st) s[tid] += s[tid + st]; __syncthreads(); }
  if (tid == 0) part[b * gridDim.x + blockIdx.x] = s[0];
}

__global__ void k_nll_fin(const float* __restrict__ part, const float* __restrict__ acc_ld,
                          const float* __restrict__ logq, float* __restrict__ out, int n) {
  int b = threadIdx.x;
  if (b >= B_) return;
  float s = 0.f;
  for (int i = 0; i < n; ++i) s += part[b * n + i];
  out[b] = (s - acc_ld[b]) + logq[b];
}

// ----------------------------- host side ----------------------------------
struct DdsP { const float *b1, *b2, *g1, *g2, *pw_b, *pw_w, *sep_b, *sep_w; };
struct CfP  { DdsP dds[3]; const float *pre_b, *pre_w, *proj_b, *proj_w; };

} // anonymous namespace

extern "C" void kernel_launch(void* const* d_in, const int* in_sizes, int n_in,
                              void* d_out, int out_size, void* d_ws, size_t ws_size,
                              hipStream_t stream) {
  (void)in_sizes; (void)n_in; (void)out_size; (void)ws_size;

  auto gp = [&](int k) { return (const float*)d_in[k]; };
  // setup_inputs() insertion order: x, w, e_q, main_params, post_params,
  // post_flow_params, flow_params.  Param pytrees flattened jax-style
  // (dict keys sorted alphabetically).
  const float* w_in = gp(1);
  const float* e_q  = gp(2);
  int idx = 3;
  idx += 28;  // main_params (unused): dds(3x8) + pre_b, pre_w, proj_b, proj_w
  idx += 28;  // post_params (unused)

  auto parse_cf = [&](CfP& cf) {
    for (int l = 0; l < 3; ++l) {   // dds layer dict keys sorted
      DdsP& d = cf.dds[l];
      d.b1 = gp(idx++); d.b2 = gp(idx++); d.g1 = gp(idx++); d.g2 = gp(idx++);
      d.pw_b = gp(idx++); d.pw_w = gp(idx++); d.sep_b = gp(idx++); d.sep_w = gp(idx++);
    }
    cf.pre_b = gp(idx++); cf.pre_w = gp(idx++); cf.proj_b = gp(idx++); cf.proj_w = gp(idx++);
  };

  CfP pf[4], ff[4];
  for (int f = 0; f < 4; ++f) parse_cf(pf[f]);          // post_flow: convflows < logs < m
  const float* pf_logs = gp(idx++);
  const float* pf_m    = gp(idx++);
  for (int f = 0; f < 4; ++f) parse_cf(ff[f]);          // flow_params
  const float* f_logs = gp(idx++);
  const float* f_m    = gp(idx++);

  // workspace carve-up
  char* p = (char*)d_ws;
  auto carve = [&](size_t bytes) { void* r = p; p += (bytes + 255) & ~(size_t)255; return r; };
  float*          z     = (float*)carve((size_t)B_ * 2 * T_ * 4);
  float*          h0    = (float*)carve((size_t)B_ * C_ * T_ * 4);
  float*          h1    = (float*)carve((size_t)B_ * C_ * T_ * 4);
  unsigned short* wpw   = (unsigned short*)carve((size_t)3 * C_ * C_ * 2);
  unsigned short* wpr   = (unsigned short*)carve((size_t)32 * C_ * 2);
  float*          wprb  = (float*)carve(32 * 4);
  float*          p_lad = (float*)carve((size_t)B_ * 64 * 4);
  float*          p0    = (float*)carve((size_t)B_ * 8 * 4);
  float*          p1    = (float*)carve((size_t)B_ * 8 * 4);
  float*          p2    = (float*)carve((size_t)B_ * 8 * 4);
  float*          accld = (float*)carve(B_ * 4);
  float*          logq  = (float*)carve(B_ * 4);

  const dim3 gTile(T_ / 32, B_);
  const int  nPrep = (3 * C_ * C_ + 32 * C_ + 32 + 255) / 256;
  const int  nZ    = (B_ * 2 * T_ + 255) / 256;

  auto conv_flow = [&](const CfP& cf) {
    k_prep<<<nPrep, 256, 0, stream>>>(cf.dds[0].pw_w, cf.dds[1].pw_w, cf.dds[2].pw_w,
                                      cf.proj_w, cf.proj_b, wpw, wpr, wprb);
    // layer 0 fuses the pre 1x1 expansion (reads z directly, no 48MB h round trip)
    k_dds<1, true><<<gTile, 384, 0, stream>>>(z, h1, cf.dds[0].sep_w, cf.dds[0].sep_b,
                                              cf.dds[0].g1, cf.dds[0].b1, wpw, cf.dds[0].pw_b,
                                              cf.dds[0].g2, cf.dds[0].b2, cf.pre_w, cf.pre_b);
    k_dds<3, false><<<gTile, 384, 0, stream>>>(h1, h0, cf.dds[1].sep_w, cf.dds[1].sep_b,
                                               cf.dds[1].g1, cf.dds[1].b1, wpw + C_ * C_, cf.dds[1].pw_b,
                                               cf.dds[1].g2, cf.dds[1].b2, cf.pre_w, cf.pre_b);
    k_dds<9, false><<<gTile, 384, 0, stream>>>(h0, h1, cf.dds[2].sep_w, cf.dds[2].sep_b,
                                               cf.dds[2].g1, cf.dds[2].b1, wpw + 2 * C_ * C_, cf.dds[2].pw_b,
                                               cf.dds[2].g2, cf.dds[2].b2, cf.pre_w, cf.pre_b);
    k_proj_rqs<<<gTile, 256, 0, stream>>>(h1, z, wpr, wprb, p_lad);
    k_reduce_lad<<<B_, 64, 0, stream>>>(p_lad, accld, 64);
  };

  // chain 1: post_flow on e_q
  k_flow_init<<<nZ, 256, 0, stream>>>(e_q, z, pf_m, pf_logs);
  k_acc_logs<<<1, B_, 0, stream>>>(accld, pf_logs, 0);
  for (int f = 0; f < 4; ++f) conv_flow(pf[f]);

  // middle: sigmoid / logq / y0
  k_mid_elem<<<dim3(T_ / 256, B_), 256, 0, stream>>>(z, w_in, e_q, p0, p1, p2);
  k_mid_fin<<<1, B_, 0, stream>>>(p0, p1, p2, accld, logq, T_ / 256);

  // chain 2: flow on [y0, z1]  (in-place affine init)
  k_flow_init<<<nZ, 256, 0, stream>>>(z, z, f_m, f_logs);
  k_acc_logs<<<1, B_, 0, stream>>>(accld, f_logs, 1);
  for (int f = 0; f < 4; ++f) conv_flow(ff[f]);

  // final NLL
  k_nll_elem<<<dim3(T_ / 256, B_), 256, 0, stream>>>(z, p0);
  k_nll_fin<<<1, B_, 0, stream>>>(p0, accld, logq, (float*)d_out, T_ / 256);
}
